// SegmentationModelMultiHead_59536836657330
// MI455X (gfx1250) — matrix operand order repaired and checked by reading, not compiled, not run on hardware
//
#include <hip/hip_runtime.h>
#include <hip/hip_bf16.h>

typedef unsigned int   u32;
typedef unsigned short u16;

typedef __attribute__((ext_vector_type(16))) __bf16 bf16x16;
typedef __attribute__((ext_vector_type(8)))  float  f32x8;

union Frag16 {           // 16 bf16 = 32 bytes = one WMMA A or B fragment per lane
    bf16x16 v;
    uint4   q[2];
};

__device__ __forceinline__ u16 f32_to_bf16(float f) {
    u32 u = __float_as_uint(f);
    u32 r = u + 0x7fffu + ((u >> 16) & 1u);   // round-to-nearest-even
    return (u16)(r >> 16);
}
__device__ __forceinline__ float bf16_to_f32(u16 h) {
    return __uint_as_float(((u32)h) << 16);
}

// ---------------------------------------------------------------------------
// Kernel 0: pack dec_w [64][64][3][3] f32 (OIHW) into WMMA B-fragment layout:
//   Wd[nt][t][h][lane][k]  (bf16), nt=N-tile(4), t=tap(9), h=K-half(2)
//   lane<16 : N=lane,    Klocal = k        (k = 0..15)
//   lane>=16: N=lane-16, Klocal = 16 + k
//   channel = 32*h + Klocal
// ---------------------------------------------------------------------------
__global__ void pack_dec_w_kernel(const float* __restrict__ dec_w,
                                  u16* __restrict__ Wd) {
    int i = blockIdx.x * blockDim.x + threadIdx.x;
    const int total = 4 * 9 * 2 * 32 * 16;
    if (i >= total) return;
    int k    = i & 15;   int rest = i >> 4;
    int lane = rest & 31; rest >>= 5;
    int h    = rest & 1;  rest >>= 1;
    int t    = rest % 9;
    int nt   = rest / 9;
    int Nl   = lane & 15;
    int Kl   = (lane < 16) ? k : (16 + k);
    int ch   = 32 * h + Kl;
    int n    = nt * 16 + Nl;
    Wd[i] = f32_to_bf16(dec_w[(n * 64 + ch) * 9 + t]);
}

// ---------------------------------------------------------------------------
// Kernel 1: encoder. x [16][3][512][512] f32 NCHW, stride-2 3x3 SAME
// (pad_lo=0, pad_hi=1 -> iy = 2y+ty, valid iff < 512), +bias, ReLU.
// Output f: bf16 NHWC [16][256][256][64].
// One thread computes all 64 output channels of one pixel (weights from LDS,
// uniform address -> LDS broadcast).
// ---------------------------------------------------------------------------
__global__ void __launch_bounds__(256, 1)
enc_kernel(const float* __restrict__ x,
           const float* __restrict__ enc_w,
           const float* __restrict__ enc_b,
           u16* __restrict__ f) {
    __shared__ float w_s[64 * 27];
    __shared__ float b_s[64];
    for (int i = threadIdx.x; i < 64 * 27; i += blockDim.x) w_s[i] = enc_w[i];
    if (threadIdx.x < 64) b_s[threadIdx.x] = enc_b[threadIdx.x];
    __syncthreads();

    int pix = blockIdx.x * blockDim.x + threadIdx.x;   // [0, 16*256*256)
    int xo = pix & 255;
    int yo = (pix >> 8) & 255;
    int b  = pix >> 16;

    float acc[64];
#pragma unroll
    for (int o = 0; o < 64; ++o) acc[o] = 0.0f;

    for (int c = 0; c < 3; ++c) {
        for (int ty = 0; ty < 3; ++ty) {
            int iy = 2 * yo + ty;
            if (iy >= 512) continue;
            for (int tx = 0; tx < 3; ++tx) {
                int ix = 2 * xo + tx;
                if (ix >= 512) continue;
                float v = x[((b * 3 + c) * 512 + iy) * 512 + ix];
                const float* wp = &w_s[c * 9 + ty * 3 + tx];
#pragma unroll
                for (int o = 0; o < 64; ++o)
                    acc[o] = fmaf(v, wp[o * 27], acc[o]);
            }
        }
    }

    u32 packed[32];
#pragma unroll
    for (int p = 0; p < 32; ++p) {
        float v0 = acc[2 * p]     + b_s[2 * p];
        float v1 = acc[2 * p + 1] + b_s[2 * p + 1];
        v0 = v0 > 0.0f ? v0 : 0.0f;
        v1 = v1 > 0.0f ? v1 : 0.0f;
        packed[p] = (u32)f32_to_bf16(v0) | ((u32)f32_to_bf16(v1) << 16);
    }
    uint4* dst = (uint4*)(f + (size_t)pix * 64);
#pragma unroll
    for (int j = 0; j < 8; ++j) dst[j] = ((uint4*)packed)[j];
}

// ---------------------------------------------------------------------------
// Kernel 2: decoder implicit-GEMM conv 64->64, 3x3 stride-1 SAME, ReLU.
// M = pixels (16-wide x strips), N = 64 out channels (4x 16 tiles),
// K = 9 taps * 64 ch = 576 -> 18 WMMA(16x16x32 bf16) steps, fused into one
// ROLLED kk loop (kk = tap*2 + half) to keep live VGPRs ~130 and avoid the
// scratch spills seen with full unroll.
// Each wave: 2 M-tiles (rows y0,y0+1, same x range) x 4 N-tiles = 8 acc tiles.
// A-fragment (contiguous k order, matching packed B): per lane two 16B chunks of the pixel's
// NHWC channel run:  lanes 0-15 chunks at ch {32h+0, 32h+8},
//                    lanes 16-31 chunks at ch {32h+16, 32h+24}.
// Padding handled by clamped addresses + zero cndmask (EXEC stays all ones).
// ---------------------------------------------------------------------------
__global__ void __launch_bounds__(256, 1)
dec_kernel(const u16* __restrict__ f,
           const u16* __restrict__ Wd,
           const float* __restrict__ dec_b,
           u16* __restrict__ d) {
    const int lane   = threadIdx.x & 31;
    const int wid    = blockIdx.x * (blockDim.x >> 5) + (threadIdx.x >> 5); // [0,32768)
    const int b      = wid >> 11;          // / (16 * 128)
    const int r      = wid & 2047;
    const int x0     = (r >> 7) << 4;      // x-strip start
    const int y0     = (r & 127) << 1;     // row pair

    const int lane16 = lane & 15;
    const int hi     = lane >> 4;          // 0 | 1
    const int px0    = x0 + lane16;        // this lane's M-row pixel (x)
    const int choff  = hi << 4;            // lanes>=16: +16 channels

    // Per-lane invariants hoisted out of the K loop.
    const u16* fb = f + ((size_t)b << 22);              // sample base (256*256*64)
    const u16* wb = Wd + (lane << 4);                   // lane's B-frag base

    f32x8 acc[2][4];
    const f32x8 z8 = {0.f, 0.f, 0.f, 0.f, 0.f, 0.f, 0.f, 0.f};
#pragma unroll
    for (int m = 0; m < 2; ++m)
#pragma unroll
        for (int nt = 0; nt < 4; ++nt) acc[m][nt] = z8;

    uint4 zq; zq.x = zq.y = zq.z = zq.w = 0u;

#pragma unroll 1
    for (int kk = 0; kk < 18; ++kk) {
        const int t  = kk >> 1;            // tap 0..8
        const int h  = kk & 1;             // K half 0..1
        const int ty = t / 3, tx = t % 3;

        const int px   = px0 + tx - 1;
        const bool pxv = (px >= 0) && (px < 256);
        const int pxc  = px < 0 ? 0 : (px > 255 ? 255 : px);

        // B fragments: one 32B contiguous run per lane per N-tile.
        Frag16 bf[4];
#pragma unroll
        for (int nt = 0; nt < 4; ++nt) {
            const uint4* wp = (const uint4*)(wb + ((nt * 18 + kk) << 9));
            bf[nt].q[0] = wp[0];
            bf[nt].q[1] = wp[1];
        }

#pragma unroll
        for (int m = 0; m < 2; ++m) {
            const int py  = y0 + m + ty - 1;
            const bool ok = pxv && (py >= 0) && (py < 256);
            const int pyc = py < 0 ? 0 : (py > 255 ? 255 : py);
            const uint4* ap = (const uint4*)(
                fb + (size_t)((pyc << 8) + pxc) * 64 + (h << 5) + choff);
            uint4 t0 = ap[0];     // channels K..K+7
            uint4 t1 = ap[1];     // channels K+8..K+15 (16B further)
            Frag16 a;
            a.q[0] = ok ? t0 : zq;
            a.q[1] = ok ? t1 : zq;
#pragma unroll
            for (int nt = 0; nt < 4; ++nt)
                acc[m][nt] = __builtin_amdgcn_wmma_f32_16x16x32_bf16(
                    false, a.v, false, bf[nt].v, (short)0, acc[m][nt],
                    false, false);
        }
    }

    // Epilogue: D layout -> VGPR r8: lanes 0-15 = (M=r8, N=lane),
    //                                 lanes 16-31 = (M=8+r8, N=lane-16).
#pragma unroll
    for (int m = 0; m < 2; ++m) {
        const int y = y0 + m;
#pragma unroll
        for (int nt = 0; nt < 4; ++nt) {
            const int c    = nt * 16 + lane16;
            const float bv = dec_b[c];
#pragma unroll
            for (int r8 = 0; r8 < 8; ++r8) {
                const int px2 = x0 + r8 + (hi << 3);
                float v = acc[m][nt][r8] + bv;
                v = v > 0.0f ? v : 0.0f;
                d[(size_t)((((b << 8) + y) << 8) + px2) * 64 + c] = f32_to_bf16(v);
            }
        }
    }
}

// ---------------------------------------------------------------------------
// Kernel 3: routed head. Per sample b, weights head_w[cls_id[b]] (1x64x3x3)
// cached in LDS; each thread computes one output pixel (576-term dot, bf16
// activations unpacked from 16B vector loads). Output f32 [16][1][256][256].
// ---------------------------------------------------------------------------
__global__ void head_kernel(const u16* __restrict__ d,
                            const float* __restrict__ head_w,
                            const float* __restrict__ head_b,
                            const int* __restrict__ cls_id,
                            float* __restrict__ out) {
    __shared__ float w_s[576];
    __shared__ float bias_s;
    const int bid  = blockIdx.x;
    const int b    = bid >> 8;
    const int tile = bid & 255;
    const int tx0  = (tile & 15) << 4;
    const int ty0  = (tile >> 4) << 4;
    const int cls  = cls_id[b];
    for (int i = threadIdx.x; i < 576; i += blockDim.x)
        w_s[i] = head_w[cls * 576 + i];            // [cls][0][ch][3][3]
    if (threadIdx.x == 0) bias_s = head_b[cls];    // n_classes = 1
    __syncthreads();

    const int xo = tx0 + (threadIdx.x & 15);
    const int yo = ty0 + (threadIdx.x >> 4);

    float acc = bias_s;
    for (int ty = 0; ty < 3; ++ty) {
        const int py = yo + ty - 1;
        if ((unsigned)py >= 256u) continue;
        for (int tx = 0; tx < 3; ++tx) {
            const int px = xo + tx - 1;
            if ((unsigned)px >= 256u) continue;
            const int t = ty * 3 + tx;
            const u16* dp = d + (size_t)((((b << 8) + py) << 8) + px) * 64;
#pragma unroll
            for (int cq = 0; cq < 8; ++cq) {
                uint4 q = *(const uint4*)(dp + cq * 8);
                const u32 w[4] = {q.x, q.y, q.z, q.w};
#pragma unroll
                for (int j = 0; j < 4; ++j) {
                    const int ch = cq * 8 + 2 * j;
                    acc = fmaf(bf16_to_f32((u16)(w[j] & 0xffffu)),
                               w_s[ch * 9 + t], acc);
                    acc = fmaf(bf16_to_f32((u16)(w[j] >> 16)),
                               w_s[(ch + 1) * 9 + t], acc);
                }
            }
        }
    }
    out[(size_t)((b << 8) + yo) * 256 + xo] = acc;
}

// ---------------------------------------------------------------------------
// Host-side launcher
// ---------------------------------------------------------------------------
extern "C" void kernel_launch(void* const* d_in, const int* in_sizes, int n_in,
                              void* d_out, int out_size, void* d_ws, size_t ws_size,
                              hipStream_t stream) {
    const float* x      = (const float*)d_in[0];   // [16,3,512,512]
    const int*   cls_id = (const int*)d_in[1];     // [16]
    const float* enc_w  = (const float*)d_in[2];   // [64,3,3,3]
    const float* enc_b  = (const float*)d_in[3];   // [64]
    const float* dec_w  = (const float*)d_in[4];   // [64,64,3,3]
    const float* dec_b  = (const float*)d_in[5];   // [64]
    const float* head_w = (const float*)d_in[6];   // [5,1,64,3,3]
    const float* head_b = (const float*)d_in[7];   // [5,1]
    float* out = (float*)d_out;                    // [16,1,256,256] f32

    char* ws = (char*)d_ws;
    const size_t FEAT_BYTES = (size_t)16 * 256 * 256 * 64 * sizeof(u16); // 128 MiB
    u16* f  = (u16*)(ws);                          // encoder out, bf16 NHWC
    u16* dd = (u16*)(ws + FEAT_BYTES);             // decoder out, bf16 NHWC
    u16* Wd = (u16*)(ws + 2 * FEAT_BYTES);         // packed decoder weights

    (void)in_sizes; (void)n_in; (void)out_size; (void)ws_size;

    pack_dec_w_kernel<<<144, 256, 0, stream>>>(dec_w, Wd);
    enc_kernel<<<(16 * 256 * 256) / 256, 256, 0, stream>>>(x, enc_w, enc_b, f);
    dec_kernel<<<4096, 256, 0, stream>>>(f, Wd, dec_b, dd);
    head_kernel<<<4096, 256, 0, stream>>>(dd, head_w, head_b, cls_id, out);
}
